// MSRF_79980880986681
// MI455X (gfx1250) — matrix operand order repaired and checked by reading, not compiled, not run on hardware
//
#include <hip/hip_runtime.h>

// Mamba block for MI455X (gfx1250, wave32, WMMA).
// B=4, L=4096, D_MODEL=128, D_INNER=256, D_STATE=16, DT_RANK=8, D_CONV=4
#define B_    4
#define L_    4096
#define DM    128
#define DI    256
#define DS    16
#define DTR   8
#define DCONV 4

typedef __attribute__((ext_vector_type(16))) __bf16        v16bf;
typedef __attribute__((ext_vector_type(8)))  float         v8f;
typedef __attribute__((ext_vector_type(4)))  unsigned int  v4u;

__device__ __forceinline__ unsigned short f2bf(float f) {
  unsigned int u = __float_as_uint(f);
  u += 0x7FFFu + ((u >> 16) & 1u);          // round-to-nearest-even
  return (unsigned short)(u >> 16);
}

// ---------------- 1) RMSNorm + cast to bf16 (one wave32 per row of 128) ----
__global__ void k_rmsnorm(const float* __restrict__ x, const float* __restrict__ w,
                          unsigned short* __restrict__ h) {
  const int row  = blockIdx.x * 8 + (threadIdx.x >> 5);
  const int lane = threadIdx.x & 31;
  const float4 v = reinterpret_cast<const float4*>(x + (size_t)row * DM)[lane];
  float ss = v.x * v.x + v.y * v.y + v.z * v.z + v.w * v.w;
#pragma unroll
  for (int off = 16; off > 0; off >>= 1) ss += __shfl_xor(ss, off, 32);
  const float sc = rsqrtf(ss * (1.0f / DM) + 1e-5f);
  const float4 wv = reinterpret_cast<const float4*>(w)[lane];
  ushort4 o;
  o.x = f2bf(v.x * sc * wv.x);
  o.y = f2bf(v.y * sc * wv.y);
  o.z = f2bf(v.z * sc * wv.z);
  o.w = f2bf(v.w * sc * wv.w);
  reinterpret_cast<ushort4*>(h + (size_t)row * DM)[lane] = o;
}

// ---------------- 2) Convert + transpose weights to bf16 (N-major) --------
__global__ void k_convert_w(const float* __restrict__ inW,   // 128 x 512
                            const float* __restrict__ outW,  // 256 x 128
                            unsigned short* __restrict__ WAt, // 512 x 128
                            unsigned short* __restrict__ WOt) // 128 x 256
{
  const int idx = blockIdx.x * 256 + threadIdx.x;
  if (idx < DM * 2 * DI) {
    const int k = idx / (2 * DI), n = idx % (2 * DI);
    WAt[n * DM + k] = f2bf(inW[idx]);
  }
  if (idx < DI * DM) {
    const int k = idx / DM, n = idx % DM;
    WOt[n * DI + k] = f2bf(outW[idx]);
  }
}

// ---------------- WMMA bf16 GEMM: C[M,N] = A[M,K] * Bt[N,K]^T (+resid) ----
// A fragment (16x32, 16-bit): lane<16 -> K = kk+0..15; lane>=16 -> kk+16..31
__device__ __forceinline__ v16bf load_a_frag(const unsigned short* p) {
  union { v16bf v; v4u u[2]; } f;
  f.u[0] = *reinterpret_cast<const v4u*>(p);       // K = +0..7
  f.u[1] = *reinterpret_cast<const v4u*>(p + 8);   // K = +8..15
  return f.v;
}
// B fragment (32x16, 16-bit): lane<16 -> K = kk+0..15, lane>=16 -> kk+16..31
__device__ __forceinline__ v16bf load_b_frag(const unsigned short* p) {
  union { v16bf v; v4u u[2]; } f;
  f.u[0] = *reinterpret_cast<const v4u*>(p);
  f.u[1] = *reinterpret_cast<const v4u*>(p + 8);
  return f.v;
}

__global__ void k_wmma_gemm(const unsigned short* __restrict__ A,   // M x K bf16 row-major
                            const unsigned short* __restrict__ Bt,  // N x K bf16 (B col-major)
                            const float* __restrict__ resid,        // M x N or nullptr
                            float* __restrict__ C,                  // M x N f32
                            int K, int N) {
  const int lane   = threadIdx.x & 31;
  const int lane16 = lane & 15;
  const int laneHi = lane >> 4;
  const int wave   = threadIdx.x >> 5;
  const int aRowId = blockIdx.x * 128 + wave * 16 + lane16;
  const int col0   = blockIdx.y * 64;

  const unsigned short* aRow = A  + (size_t)aRowId * K;
  const unsigned short* b0   = Bt + (size_t)(col0 + 0 * 16 + lane16) * K;
  const unsigned short* b1   = Bt + (size_t)(col0 + 1 * 16 + lane16) * K;
  const unsigned short* b2   = Bt + (size_t)(col0 + 2 * 16 + lane16) * K;
  const unsigned short* b3   = Bt + (size_t)(col0 + 3 * 16 + lane16) * K;

  v8f acc0 = {}, acc1 = {}, acc2 = {}, acc3 = {};
  for (int kk = 0; kk < K; kk += 32) {
    __builtin_prefetch(aRow + kk + 32, 0, 0);        // global_prefetch_b8
    const v16bf a  = load_a_frag(aRow + kk + laneHi * 16);
    const v16bf f0 = load_b_frag(b0 + kk + laneHi * 16);
    const v16bf f1 = load_b_frag(b1 + kk + laneHi * 16);
    const v16bf f2 = load_b_frag(b2 + kk + laneHi * 16);
    const v16bf f3 = load_b_frag(b3 + kk + laneHi * 16);
    acc0 = __builtin_amdgcn_wmma_f32_16x16x32_bf16(false, a, false, f0, (short)0, acc0, false, false);
    acc1 = __builtin_amdgcn_wmma_f32_16x16x32_bf16(false, a, false, f1, (short)0, acc1, false, false);
    acc2 = __builtin_amdgcn_wmma_f32_16x16x32_bf16(false, a, false, f2, (short)0, acc2, false, false);
    acc3 = __builtin_amdgcn_wmma_f32_16x16x32_bf16(false, a, false, f3, (short)0, acc3, false, false);
  }

  // C/D layout: VGPR r -> row (r + 8*laneHi), col = lane16
  const int rowOut = blockIdx.x * 128 + wave * 16 + laneHi * 8;
  v8f accs[4] = {acc0, acc1, acc2, acc3};
#pragma unroll
  for (int j = 0; j < 4; ++j) {
    const int col = col0 + j * 16 + lane16;
#pragma unroll
    for (int r = 0; r < 8; ++r) {
      const size_t off = (size_t)(rowOut + r) * N + col;
      float v = accs[j][r];
      if (resid) v += resid[off];
      C[off] = v;
    }
  }
}

// ---------------- 3) causal depthwise conv(4) + SiLU ----------------------
__global__ void k_conv_silu(const float* __restrict__ xr, const float* __restrict__ cw,
                            const float* __restrict__ cb, float* __restrict__ xc) {
  const size_t idx = (size_t)blockIdx.x * 256 + threadIdx.x;  // over B*L*DI
  const int d  = (int)(idx & (DI - 1));
  const size_t bl = idx >> 8;                                 // b*L + l
  const int l  = (int)(bl & (L_ - 1));
  float acc = cb[d];
#pragma unroll
  for (int j = 0; j < DCONV; ++j) {
    const int ll = l - (DCONV - 1) + j;
    if (ll >= 0) acc += cw[d * DCONV + j] * xr[(bl - (DCONV - 1) + j) * (2 * DI) + d];
  }
  xc[idx] = acc / (1.0f + __expf(-acc));  // silu
}

// ---------------- 4) x_proj (K=256,N=40) + dt_proj + softplus --------------
__global__ void k_xproj(const float* __restrict__ xc, const float* __restrict__ xpw,
                        const float* __restrict__ dtw, const float* __restrict__ dtb,
                        float* __restrict__ delta, float* __restrict__ Bm,
                        float* __restrict__ Cm) {
  __shared__ float xs[DI];
  __shared__ float xd[DTR + 2 * DS];   // 40
  const size_t row = blockIdx.x;
  const int tid = threadIdx.x;
  xs[tid] = xc[row * DI + tid];
  __syncthreads();
  if (tid < DTR + 2 * DS) {
    float acc = 0.0f;
    for (int k = 0; k < DI; ++k) acc += xs[k] * xpw[k * (DTR + 2 * DS) + tid];
    xd[tid] = acc;
  }
  __syncthreads();
  float acc = dtb[tid];
#pragma unroll
  for (int r = 0; r < DTR; ++r) acc += xd[r] * dtw[r * DI + tid];
  delta[row * DI + tid] = (acc > 20.0f) ? acc : log1pf(__expf(acc));
  if (tid < DS)           Bm[row * DS + tid]        = xd[DTR + tid];
  else if (tid < 2 * DS)  Cm[row * DS + (tid - DS)] = xd[DTR + DS + (tid - DS)];
}

// ---------------- 5) selective scan + gating (lane = channel) --------------
__global__ void k_scan(const float* __restrict__ delta, const float* __restrict__ Bm,
                       const float* __restrict__ Cm, const float* __restrict__ xc,
                       const float* __restrict__ xr, const float* __restrict__ Alog,
                       const float* __restrict__ Dp, unsigned short* __restrict__ ybf) {
  __shared__ float sB[128 * DS];
  __shared__ float sC[128 * DS];
  const int b = blockIdx.x;
  const int d = threadIdx.x;
  float A[DS], s[DS];
#pragma unroll
  for (int n = 0; n < DS; ++n) { A[n] = -__expf(Alog[d * DS + n]); s[n] = 0.0f; }
  const float Dd = Dp[d];
  const size_t rowBase = (size_t)b * L_;
  for (int l0 = 0; l0 < L_; l0 += 128) {
    __syncthreads();
    for (int i = threadIdx.x; i < 128 * DS; i += 256) {
      const size_t g = (rowBase + l0) * DS + i;
      sB[i] = Bm[g];
      sC[i] = Cm[g];
    }
    __syncthreads();
    for (int t = 0; t < 128; ++t) {
      const size_t row = rowBase + l0 + t;
      const float dl  = delta[row * DI + d];
      const float u   = xc[row * DI + d];
      const float dbu = dl * u;
      float y = 0.0f;
#pragma unroll
      for (int n = 0; n < DS; ++n) {
        const float dA = __expf(dl * A[n]);
        s[n] = dA * s[n] + dbu * sB[t * DS + n];   // broadcast LDS read
        y += s[n] * sC[t * DS + n];
      }
      const float r = xr[row * (2 * DI) + DI + d];       // res half of xr
      const float silu_r = r / (1.0f + __expf(-r));
      ybf[row * DI + d] = f2bf((y + u * Dd) * silu_r);
    }
  }
}

// ---------------------------------------------------------------------------
extern "C" void kernel_launch(void* const* d_in, const int* in_sizes, int n_in,
                              void* d_out, int out_size, void* d_ws, size_t ws_size,
                              hipStream_t stream) {
  const float* x    = (const float*)d_in[0];
  const float* nw   = (const float*)d_in[1];
  const float* inW  = (const float*)d_in[2];
  const float* cw   = (const float*)d_in[3];
  const float* cb   = (const float*)d_in[4];
  const float* xpw  = (const float*)d_in[5];
  const float* dtw  = (const float*)d_in[6];
  const float* dtb  = (const float*)d_in[7];
  const float* Alog = (const float*)d_in[8];
  const float* Dp   = (const float*)d_in[9];
  const float* outW = (const float*)d_in[10];
  float* out = (float*)d_out;
  (void)in_sizes; (void)n_in; (void)out_size; (void)ws_size;

  char* ws = (char*)d_ws;
  size_t off = 0;
  auto carve = [&](size_t bytes) -> void* {
    void* p = ws + off;
    off = (off + bytes + 255) & ~(size_t)255;
    return p;
  };
  const size_t R = (size_t)B_ * L_;  // 16384 rows
  unsigned short* hbf = (unsigned short*)carve(R * DM * 2);            // 4 MB
  unsigned short* WAt = (unsigned short*)carve((size_t)2 * DI * DM * 2);
  unsigned short* WOt = (unsigned short*)carve((size_t)DM * DI * 2);
  float* xrbuf = (float*)carve(R * 2 * DI * 4);                        // 32 MB
  float* xcbuf = (float*)carve(R * DI * 4);                            // 16 MB
  float* dlt   = (float*)carve(R * DI * 4);                            // 16 MB
  float* Bmb   = (float*)carve(R * DS * 4);                            // 1 MB
  float* Cmb   = (float*)carve(R * DS * 4);                            // 1 MB
  unsigned short* ybf = (unsigned short*)carve(R * DI * 2);            // 8 MB

  // 1) RMSNorm -> bf16 activations
  k_rmsnorm<<<(unsigned)(R / 8), 256, 0, stream>>>(x, nw, hbf);
  // 2) weights -> transposed bf16
  k_convert_w<<<(DM * 2 * DI + 255) / 256, 256, 0, stream>>>(inW, outW, WAt, WOt);
  // in_proj: xr[16384,512] = hbf[16384,128] @ W  (WMMA bf16)
  k_wmma_gemm<<<dim3((unsigned)(R / 128), (2 * DI) / 64), 256, 0, stream>>>(
      hbf, WAt, nullptr, xrbuf, DM, 2 * DI);
  // 3) depthwise causal conv + silu
  k_conv_silu<<<(unsigned)((R * DI) / 256), 256, 0, stream>>>(xrbuf, cw, cb, xcbuf);
  // 4) x_proj + dt_proj + softplus
  k_xproj<<<(unsigned)R, 256, 0, stream>>>(xcbuf, xpw, dtw, dtb, dlt, Bmb, Cmb);
  // 5) selective scan + D skip + gating -> bf16 y
  k_scan<<<B_, 256, 0, stream>>>(dlt, Bmb, Cmb, xcbuf, xrbuf, Alog, Dp, ybf);
  // 6) out_proj + residual: out[16384,128] = y @ Wo + x  (WMMA bf16)
  k_wmma_gemm<<<dim3((unsigned)(R / 128), DM / 64), 256, 0, stream>>>(
      ybf, WOt, x, out, DI, DM);
}
